// AdaLN_17669495456065
// MI455X (gfx1250) — matrix operand order repaired and checked by reading, not compiled, not run on hardware
//
#include <hip/hip_runtime.h>
#include <stdint.h>

#define LN_EPS 1e-5f

typedef __attribute__((ext_vector_type(16))) _Float16 v16h;
typedef __attribute__((ext_vector_type(8)))  float    f32x8;

union HFrag {
  v16h v;
  _Float16 h[16];
  uint4 q[2];
};

__device__ __forceinline__ float silu_f(float x) {
  return x / (1.0f + __expf(-x));
}

__device__ __forceinline__ float wave_sum(float v) {
#pragma unroll
  for (int o = 16; o > 0; o >>= 1) v += __shfl_xor(v, o, 32);
  return v;
}

// A fragment: 16x32 f16 tile at column-base kb from LDS row-major [16 x KT].
// ISA layout: lanes 0-15 hold K = kb+{0..7, 16..23}; lanes 16-31 hold K = kb+{8..15, 24..31}.
__device__ __forceinline__ HFrag load_a_frag(const _Float16* Abf, int KT, int kb, int lane) {
  const int m = lane & 15;
  const int half = lane >> 4;
  HFrag a;
  const _Float16* base = Abf + m * KT + kb + half * 8;
  a.q[0] = *(const uint4*)(base);       // K = kb + half*8 .. +7
  a.q[1] = *(const uint4*)(base + 16);  // K = kb + 16 + half*8 .. +7
  return a;
}

// B fragment from PRE-CONVERTED f16 weights, row-major [NOUT x KT]; B[k][n] = W[n][k].
// ISA layout: lane n holds column n; lanes 0-15 K=kb+{0..7,16..23}, lanes 16-31 K=kb+{8..15,24..31}.
__device__ __forceinline__ HFrag load_b_frag_h(const _Float16* __restrict__ Wh,
                                               int KT, int j, int kb, int half) {
  HFrag b;
  const uint4* wp = (const uint4*)(Wh + (size_t)j * KT + kb + half * 8);
  b.q[0] = wp[0];
  b.q[1] = wp[2];
  return b;
}

// ---------------------------------------------------------------------------
// K0: one-shot f32 -> f16 conversion of the three weight matrices into d_ws.
// Concatenated: [0,131072) W_v, [131072,163840) W_vpos, [163840,196608) W_e.
// ---------------------------------------------------------------------------
__global__ __launch_bounds__(256) void cvt_weights_kernel(
    const float* __restrict__ W_v, const float* __restrict__ W_vpos,
    const float* __restrict__ W_e, _Float16* __restrict__ h) {
  const int i = (blockIdx.x * 256 + threadIdx.x) * 4;
  const float* s;
  int off;
  if (i < 131072) {
    s = W_v;
    off = i;
  } else if (i < 163840) {
    s = W_vpos;
    off = i - 131072;
  } else {
    s = W_e;
    off = i - 163840;
  }
  const float4 v = *(const float4*)(s + off);
  _Float16* d = h + i;
  d[0] = (_Float16)v.x;
  d[1] = (_Float16)v.y;
  d[2] = (_Float16)v.z;
  d[3] = (_Float16)v.w;
}

// ---------------------------------------------------------------------------
// K1: fused node AdaLN.  mod_v = silu(fv_c+fv_pos_c) @ W_v.T + b_v
//     fv_out = LN(fv) * (1+scale_v) + shift_v
// Block: 512 threads (16 waves) handles one 16-row tile; wave w covers cols [32w,32w+32).
// ---------------------------------------------------------------------------
__global__ __launch_bounds__(512) void node_mod_kernel(
    const float* __restrict__ fv, const float* __restrict__ fv_c,
    const float* __restrict__ fv_pos_c, const _Float16* __restrict__ hW_v,
    const float* __restrict__ b_v, float* __restrict__ fv_out) {
  __shared__ _Float16 Abf[16 * 256];  // 8 KB
  __shared__ float Mod[16 * 512];     // 32 KB
  const int tid = threadIdx.x;
  const int lane = tid & 31;
  const int w = tid >> 5;
  const int half = lane >> 4;
  const long r0 = (long)blockIdx.x * 16;

  // Stage A = f16(silu(fv_c + fv_pos_c)) [16 x 256]; 8 contiguous elems / thread.
  {
    const long base = r0 * 256 + (long)tid * 8;
    const float4* c4 = (const float4*)(fv_c + base);
    const float4* p4 = (const float4*)(fv_pos_c + base);
    const int o = tid * 8;
#pragma unroll
    for (int q = 0; q < 2; ++q) {
      float4 cv = c4[q], pv = p4[q];
      Abf[o + q * 4 + 0] = (_Float16)silu_f(cv.x + pv.x);
      Abf[o + q * 4 + 1] = (_Float16)silu_f(cv.y + pv.y);
      Abf[o + q * 4 + 2] = (_Float16)silu_f(cv.z + pv.z);
      Abf[o + q * 4 + 3] = (_Float16)silu_f(cv.w + pv.w);
    }
  }
  __syncthreads();

  // GEMM: K=256, two 16-col accumulators per wave.
  f32x8 acc0 = {};
  f32x8 acc1 = {};
  const int j0 = w * 32 + (lane & 15);
#pragma unroll
  for (int kb = 0; kb < 256; kb += 32) {
    if (kb + 32 < 256)
      __builtin_prefetch(hW_v + (size_t)j0 * 256 + kb + 32, 0, 3);
    HFrag a = load_a_frag(Abf, 256, kb, lane);
    HFrag b0 = load_b_frag_h(hW_v, 256, j0, kb, half);
    HFrag b1 = load_b_frag_h(hW_v, 256, j0 + 16, kb, half);
    acc0 = __builtin_amdgcn_wmma_f32_16x16x32_f16(false, a.v, false, b0.v,
                                                  (short)0, acc0, false, false);
    acc1 = __builtin_amdgcn_wmma_f32_16x16x32_f16(false, a.v, false, b1.v,
                                                  (short)0, acc1, false, false);
  }
  // C layout: VGPR i -> (row = i + half*8, col = lane&15 within tile)
#pragma unroll
  for (int i = 0; i < 8; ++i) {
    const int row = i + half * 8;
    const int col = w * 32 + (lane & 15);
    Mod[row * 512 + col] = acc0[i];
    Mod[row * 512 + col + 16] = acc1[i];
  }
  __syncthreads();

  // Combine: wave w owns row w of the tile. LN over 256 elems of fv.
  {
    const long gr = r0 + w;
    float x[8], s = 0.f, ss = 0.f;
#pragma unroll
    for (int k = 0; k < 8; ++k) {
      x[k] = fv[gr * 256 + lane + 32 * k];
      s += x[k];
      ss += x[k] * x[k];
    }
    s = wave_sum(s);
    ss = wave_sum(ss);
    const float mean = s * (1.f / 256.f);
    const float var = ss * (1.f / 256.f) - mean * mean;
    const float rstd = rsqrtf(var + LN_EPS);
#pragma unroll
    for (int k = 0; k < 8; ++k) {
      const int c = lane + 32 * k;
      const float shift = Mod[w * 512 + c] + b_v[c];
      const float scale = Mod[w * 512 + 256 + c] + b_v[256 + c];
      fv_out[gr * 256 + c] = (x[k] - mean) * rstd * (1.f + scale) + shift;
    }
  }
}

// ---------------------------------------------------------------------------
// K2: fvp = fv_pos_c @ W_vpos.T + b_vpos   [65536 x 128] into workspace.
// Block: 256 threads (8 waves) per 16-row tile; wave w -> one 16-col n-tile.
// ---------------------------------------------------------------------------
__global__ __launch_bounds__(256) void vpos_proj_kernel(
    const float* __restrict__ fv_pos_c, const _Float16* __restrict__ hW_vpos,
    const float* __restrict__ b_vpos, float* __restrict__ fvp) {
  __shared__ _Float16 Abf[16 * 256];  // 8 KB
  const int tid = threadIdx.x;
  const int lane = tid & 31;
  const int w = tid >> 5;
  const int half = lane >> 4;
  const long r0 = (long)blockIdx.x * 16;

  {
    const long base = r0 * 256 + (long)tid * 16;
    const float4* s4 = (const float4*)(fv_pos_c + base);
    const int o = tid * 16;
#pragma unroll
    for (int q = 0; q < 4; ++q) {
      float4 v = s4[q];
      Abf[o + q * 4 + 0] = (_Float16)v.x;
      Abf[o + q * 4 + 1] = (_Float16)v.y;
      Abf[o + q * 4 + 2] = (_Float16)v.z;
      Abf[o + q * 4 + 3] = (_Float16)v.w;
    }
  }
  __syncthreads();

  f32x8 acc = {};
  const int j0 = w * 16 + (lane & 15);
#pragma unroll
  for (int kb = 0; kb < 256; kb += 32) {
    if (kb + 32 < 256)
      __builtin_prefetch(hW_vpos + (size_t)j0 * 256 + kb + 32, 0, 3);
    HFrag a = load_a_frag(Abf, 256, kb, lane);
    HFrag b = load_b_frag_h(hW_vpos, 256, j0, kb, half);
    acc = __builtin_amdgcn_wmma_f32_16x16x32_f16(false, a.v, false, b.v,
                                                 (short)0, acc, false, false);
  }
  const int col = w * 16 + (lane & 15);
  const float bias = b_vpos[col];
#pragma unroll
  for (int i = 0; i < 8; ++i) {
    const long row = r0 + i + half * 8;
    fvp[row * 128 + col] = acc[i] + bias;
  }
}

// ---------------------------------------------------------------------------
// K3: fused edge AdaLN.  h = silu(fe_c + fvp[src] + fvp[dst]);
//     mod_e = h @ W_e.T + b_e;  fe_out = LN(fe)*(1+scale_e)+shift_e
// Block: 256 threads (8 waves) per 16-edge tile; wave w covers cols [32w,32w+32).
// ---------------------------------------------------------------------------
__global__ __launch_bounds__(256) void edge_mod_kernel(
    const float* __restrict__ fe, const float* __restrict__ fe_c,
    const float* __restrict__ fvp, const int* __restrict__ src,
    const int* __restrict__ dst, const _Float16* __restrict__ hW_e,
    const float* __restrict__ b_e, float* __restrict__ fe_out) {
  __shared__ _Float16 Abf[16 * 128];  // 4 KB
  __shared__ float Mod[16 * 256];     // 16 KB
  const int tid = threadIdx.x;
  const int lane = tid & 31;
  const int w = tid >> 5;
  const int half = lane >> 4;
  const long e0 = (long)blockIdx.x * 16;

  // Stage A: gather + silu -> f16 LDS. thread (row = tid>>4, 8 cols).
  {
    const int row = tid >> 4;
    const int coff = (tid & 15) * 8;
    const long e = e0 + row;
    const long si = src[e];
    const long di = dst[e];
    const float4* pe = (const float4*)(fe_c + e * 128 + coff);
    const float4* ps = (const float4*)(fvp + si * 128 + coff);
    const float4* pd = (const float4*)(fvp + di * 128 + coff);
#pragma unroll
    for (int q = 0; q < 2; ++q) {
      float4 a = pe[q], b = ps[q], c = pd[q];
      const int o = row * 128 + coff + q * 4;
      Abf[o + 0] = (_Float16)silu_f(a.x + b.x + c.x);
      Abf[o + 1] = (_Float16)silu_f(a.y + b.y + c.y);
      Abf[o + 2] = (_Float16)silu_f(a.z + b.z + c.z);
      Abf[o + 3] = (_Float16)silu_f(a.w + b.w + c.w);
    }
  }
  __syncthreads();

  f32x8 acc0 = {};
  f32x8 acc1 = {};
  const int j0 = w * 32 + (lane & 15);
#pragma unroll
  for (int kb = 0; kb < 128; kb += 32) {
    if (kb + 32 < 128)
      __builtin_prefetch(hW_e + (size_t)j0 * 128 + kb + 32, 0, 3);
    HFrag a = load_a_frag(Abf, 128, kb, lane);
    HFrag b0 = load_b_frag_h(hW_e, 128, j0, kb, half);
    HFrag b1 = load_b_frag_h(hW_e, 128, j0 + 16, kb, half);
    acc0 = __builtin_amdgcn_wmma_f32_16x16x32_f16(false, a.v, false, b0.v,
                                                  (short)0, acc0, false, false);
    acc1 = __builtin_amdgcn_wmma_f32_16x16x32_f16(false, a.v, false, b1.v,
                                                  (short)0, acc1, false, false);
  }
#pragma unroll
  for (int i = 0; i < 8; ++i) {
    const int row = i + half * 8;
    const int col = w * 32 + (lane & 15);
    Mod[row * 256 + col] = acc0[i];
    Mod[row * 256 + col + 16] = acc1[i];
  }
  __syncthreads();

  // Combine: wave w owns rows w and w+8. LN over 128 elems of fe.
#pragma unroll
  for (int rr = 0; rr < 2; ++rr) {
    const int r = w + rr * 8;
    const long ge = e0 + r;
    float x[4], s = 0.f, ss = 0.f;
#pragma unroll
    for (int k = 0; k < 4; ++k) {
      x[k] = fe[ge * 128 + lane + 32 * k];
      s += x[k];
      ss += x[k] * x[k];
    }
    s = wave_sum(s);
    ss = wave_sum(ss);
    const float mean = s * (1.f / 128.f);
    const float var = ss * (1.f / 128.f) - mean * mean;
    const float rstd = rsqrtf(var + LN_EPS);
#pragma unroll
    for (int k = 0; k < 4; ++k) {
      const int c = lane + 32 * k;
      const float shift = Mod[r * 256 + c] + b_e[c];
      const float scale = Mod[r * 256 + 128 + c] + b_e[128 + c];
      fe_out[ge * 128 + c] = (x[k] - mean) * rstd * (1.f + scale) + shift;
    }
  }
}

// ---------------------------------------------------------------------------
// K4: per-graph position scale-norm + graph LayerNorm.
// Groups are contiguous blocks of 256 nodes (node_gid = i // 256).
// ---------------------------------------------------------------------------
__device__ __forceinline__ float block_sum_256(float v, float* sm) {
  v = wave_sum(v);
  const int lane = threadIdx.x & 31;
  const int wid = threadIdx.x >> 5;
  __syncthreads();  // protect sm from previous reduction
  if (lane == 0) sm[wid] = v;
  __syncthreads();
  float r = sm[0];
#pragma unroll
  for (int i = 1; i < 8; ++i) r += sm[i];
  return r;
}

__global__ __launch_bounds__(256) void group_kernel(
    const float* __restrict__ fv_pos, const float* __restrict__ fg,
    const float* __restrict__ g_gamma, const float* __restrict__ g_beta,
    const float* __restrict__ W_pos, const float* __restrict__ b_pos,
    float* __restrict__ fg_out, float* __restrict__ pv_out) {
  __shared__ float sm[8];
  const int g = blockIdx.x;
  const int t = threadIdx.x;
  const long n = (long)g * 256 + t;

  const float px = fv_pos[n * 3 + 0];
  const float py = fv_pos[n * 3 + 1];
  const float pz = fv_pos[n * 3 + 2];
  const float mx = block_sum_256(px, sm) * (1.f / 256.f);
  const float my = block_sum_256(py, sm) * (1.f / 256.f);
  const float mz = block_sum_256(pz, sm) * (1.f / 256.f);
  const float cx = px - mx, cy = py - my, cz = pz - mz;
  const float sq = block_sum_256(cx * cx + cy * cy + cz * cz, sm);
  const float rms = sqrtf(sq * (1.f / (256.f * 3.f)) + LN_EPS);

  const float fgv = fg[(long)g * 256 + t];
  const float sdot = block_sum_256(silu_f(fgv) * W_pos[t], sm) + b_pos[0];
  const float k = (1.f + sdot) / rms;
  pv_out[n * 3 + 0] = cx * k;
  pv_out[n * 3 + 1] = cy * k;
  pv_out[n * 3 + 2] = cz * k;

  const float s1 = block_sum_256(fgv, sm);
  const float s2 = block_sum_256(fgv * fgv, sm);
  const float mean = s1 * (1.f / 256.f);
  const float var = s2 * (1.f / 256.f) - mean * mean;
  fg_out[(long)g * 256 + t] =
      (fgv - mean) * rsqrtf(var + LN_EPS) * g_gamma[t] + g_beta[t];
}

// ---------------------------------------------------------------------------
extern "C" void kernel_launch(void* const* d_in, const int* in_sizes, int n_in,
                              void* d_out, int out_size, void* d_ws,
                              size_t ws_size, hipStream_t stream) {
  const float* fv = (const float*)d_in[0];        // [65536,256]
  const float* fe = (const float*)d_in[1];        // [262144,128]
  const float* fg = (const float*)d_in[2];        // [256,256]
  const float* fv_pos = (const float*)d_in[3];    // [65536,3]
  const float* fv_c = (const float*)d_in[4];      // [65536,256]
  const float* fe_c = (const float*)d_in[5];      // [262144,128]
  const float* fv_pos_c = (const float*)d_in[6];  // [65536,256]
  const float* W_v = (const float*)d_in[7];       // [512,256]
  const float* b_v = (const float*)d_in[8];       // [512]
  const float* W_vpos = (const float*)d_in[9];    // [128,256]
  const float* b_vpos = (const float*)d_in[10];   // [128]
  const float* W_e = (const float*)d_in[11];      // [256,128]
  const float* b_e = (const float*)d_in[12];      // [256]
  const float* g_gamma = (const float*)d_in[13];  // [256]
  const float* g_beta = (const float*)d_in[14];   // [256]
  const float* W_pos = (const float*)d_in[15];    // [1,256]
  const float* b_pos = (const float*)d_in[16];    // [1]
  const int* src = (const int*)d_in[17];          // [262144]
  const int* dst = (const int*)d_in[18];          // [262144]
  // d_in[19] (node_gid) implicit: groups are contiguous blocks of 256

  float* out = (float*)d_out;
  float* fv_out = out;             // 65536*256 = 16777216
  float* fe_out = out + 16777216;  // 262144*128 = 33554432
  float* fg_out = out + 50331648;  // 256*256   = 65536
  float* pv_out = out + 50397184;  // 65536*3   = 196608

  // Workspace layout: fvp [65536x128] f32 (32 MB), then f16 weights (384 KB).
  float* fvp = (float*)d_ws;
  _Float16* hW = (_Float16*)((char*)d_ws + (size_t)65536 * 128 * 4);
  _Float16* hW_v = hW;                 // 131072 halfs
  _Float16* hW_vpos = hW + 131072;     // 32768 halfs
  _Float16* hW_e = hW + 163840;        // 32768 halfs

  cvt_weights_kernel<<<196608 / 1024, 256, 0, stream>>>(W_v, W_vpos, W_e, hW);
  node_mod_kernel<<<65536 / 16, 512, 0, stream>>>(fv, fv_c, fv_pos_c, hW_v,
                                                  b_v, fv_out);
  vpos_proj_kernel<<<65536 / 16, 256, 0, stream>>>(fv_pos_c, hW_vpos, b_vpos,
                                                   fvp);
  edge_mod_kernel<<<262144 / 16, 256, 0, stream>>>(fe, fe_c, fvp, src, dst,
                                                   hW_e, b_e, fe_out);
  group_kernel<<<256, 256, 0, stream>>>(fv_pos, fg, g_gamma, g_beta, W_pos,
                                        b_pos, fg_out, pv_out);
}
